// MambaFiLM_70720931496699
// MI455X (gfx1250) — compile-verified, matrix operand order checked by reading, not run on hardware
//
#include <hip/hip_runtime.h>

#define BATCH   2
#define SEQ     2048
#define DMODEL  768
#define DINNER  1536
#define DSTATE  16
#define DTRANK  48
#define NLAYERS 4
#define NTOK    (BATCH*SEQ)

typedef __attribute__((ext_vector_type(16))) __bf16        v16bf;
typedef __attribute__((ext_vector_type(8)))  float         v8f;
typedef __attribute__((ext_vector_type(8)))  unsigned int  v8u;

// ---------------------------------------------------------------- helpers
__device__ __forceinline__ unsigned short f2bf(float f) {
  unsigned int u = __float_as_uint(f);
  u += 0x7FFFu + ((u >> 16) & 1u);          // round-to-nearest-even
  return (unsigned short)(u >> 16);
}

// Load one 16x32 bf16 WMMA fragment (A-layout; B uses the same per-lane
// layout indexed by N).  p points at this lane's row, dword units, already
// offset by (lane>>4)*4 for the K-half split.
__device__ __forceinline__ v16bf ldfrag(const unsigned int* __restrict__ p) {
  union { v8u u; v16bf b; } c;
  c.u[0] = p[0];  c.u[1] = p[1];  c.u[2] = p[2];  c.u[3] = p[3];
  c.u[4] = p[8];  c.u[5] = p[9];  c.u[6] = p[10]; c.u[7] = p[11];
  return c.b;
}

// ---------------------------------------------------------------- kernels
__global__ void cvt_f32_to_bf16(const float* __restrict__ src,
                                unsigned short* __restrict__ dst, int n) {
  int i = blockIdx.x * blockDim.x + threadIdx.x;
  if (i < n) dst[i] = f2bf(src[i]);
}

// One block per token; 256 threads reduce 768 squares, emit bf16 xn.
__global__ __launch_bounds__(256)
void rmsnorm_kernel(const float* __restrict__ h, const float* __restrict__ w,
                    unsigned short* __restrict__ out) {
  __shared__ float red[256];
  const int tok = blockIdx.x;
  const float* hr = h + (size_t)tok * DMODEL;
  float s = 0.f;
  for (int i = threadIdx.x; i < DMODEL; i += 256) { float v = hr[i]; s += v * v; }
  red[threadIdx.x] = s;
  __syncthreads();
  for (int o = 128; o > 0; o >>= 1) {
    if (threadIdx.x < o) red[threadIdx.x] += red[threadIdx.x + o];
    __syncthreads();
  }
  const float scale = rsqrtf(red[0] * (1.0f / DMODEL) + 1e-5f);
  for (int i = threadIdx.x; i < DMODEL; i += 256)
    out[(size_t)tok * DMODEL + i] = f2bf(hr[i] * scale * w[i]);
}

// C[M,N] = A[M,K] * W[N,K]^T  (+ resid).  bf16 in, f32 accumulate.
// Wave computes a 32(M)x64(N) tile: 2 A frags + 4 B frags -> 8 WMMAs per
// K-step of 32 (0.75 fragment-loads per WMMA; data is L2-resident).
__global__ __launch_bounds__(256)
void wmma_gemm_bf16(const unsigned short* __restrict__ Abf,
                    const unsigned short* __restrict__ Wbf,
                    float* __restrict__ C, const float* __restrict__ resid,
                    int N, int K) {
  const int lane = threadIdx.x & 31;
  const int wid  = blockIdx.x * (blockDim.x >> 5) + (threadIdx.x >> 5);
  const int nT64 = N >> 6;
  const int mt   = wid / nT64;
  const int nt   = wid - mt * nT64;
  const int row0 = mt << 5;                      // 32-row M tile
  const int n0   = nt << 6;                      // 64-col N tile
  const int ldA  = K >> 1;                       // row stride in dwords
  const int r    = lane & 15;
  const int hf4  = (lane >> 4) << 2;             // K-half offset (dwords)

  const unsigned int* Adw = (const unsigned int*)Abf;
  const unsigned int* Wdw = (const unsigned int*)Wbf;
  const unsigned int* ap0 = Adw + (size_t)(row0 +  0 + r) * ldA + hf4;
  const unsigned int* ap1 = Adw + (size_t)(row0 + 16 + r) * ldA + hf4;
  const unsigned int* wp0 = Wdw + (size_t)(n0 +  0 + r) * ldA + hf4;
  const unsigned int* wp1 = Wdw + (size_t)(n0 + 16 + r) * ldA + hf4;
  const unsigned int* wp2 = Wdw + (size_t)(n0 + 32 + r) * ldA + hf4;
  const unsigned int* wp3 = Wdw + (size_t)(n0 + 48 + r) * ldA + hf4;

  v8f a00 = {}, a01 = {}, a02 = {}, a03 = {};
  v8f a10 = {}, a11 = {}, a12 = {}, a13 = {};
  for (int k = 0; k < K; k += 32) {
    const int kdw = k >> 1;
    v16bf aA = ldfrag(ap0 + kdw);
    v16bf aB = ldfrag(ap1 + kdw);
    v16bf b0 = ldfrag(wp0 + kdw);
    v16bf b1 = ldfrag(wp1 + kdw);
    v16bf b2 = ldfrag(wp2 + kdw);
    v16bf b3 = ldfrag(wp3 + kdw);
    __builtin_prefetch(ap0 + kdw + 32, 0, 1);    // global_prefetch_b8
    __builtin_prefetch(wp0 + kdw + 32, 0, 1);
    __builtin_prefetch(wp2 + kdw + 32, 0, 1);
    a00 = __builtin_amdgcn_wmma_f32_16x16x32_bf16(false, aA, false, b0, (short)0, a00, false, false);
    a01 = __builtin_amdgcn_wmma_f32_16x16x32_bf16(false, aA, false, b1, (short)0, a01, false, false);
    a02 = __builtin_amdgcn_wmma_f32_16x16x32_bf16(false, aA, false, b2, (short)0, a02, false, false);
    a03 = __builtin_amdgcn_wmma_f32_16x16x32_bf16(false, aA, false, b3, (short)0, a03, false, false);
    a10 = __builtin_amdgcn_wmma_f32_16x16x32_bf16(false, aB, false, b0, (short)0, a10, false, false);
    a11 = __builtin_amdgcn_wmma_f32_16x16x32_bf16(false, aB, false, b1, (short)0, a11, false, false);
    a12 = __builtin_amdgcn_wmma_f32_16x16x32_bf16(false, aB, false, b2, (short)0, a12, false, false);
    a13 = __builtin_amdgcn_wmma_f32_16x16x32_bf16(false, aB, false, b3, (short)0, a13, false, false);
  }

  // C/D layout: VGPR i -> row (rowBase + (lane>>4)*8 + i), col n0+j*16+(lane&15)
  const int half = lane >> 4;
  const int col0 = n0 + r;
  v8f accs[2][4] = {{a00, a01, a02, a03}, {a10, a11, a12, a13}};
#pragma unroll
  for (int mh = 0; mh < 2; ++mh) {
    const int rb = row0 + mh * 16 + half * 8;
#pragma unroll
    for (int j = 0; j < 4; ++j) {
#pragma unroll
      for (int i = 0; i < 8; ++i) {
        size_t idx = (size_t)(rb + i) * N + col0 + j * 16;
        float v = accs[mh][j][i];
        C[idx] = resid ? (resid[idx] + v) : v;
      }
    }
  }
}

// Depthwise causal conv (K=4) + bias + SiLU on the xi half of xz.
__global__ void conv_silu_kernel(const float* __restrict__ xz,
                                 const float* __restrict__ cw,
                                 const float* __restrict__ cb,
                                 float* __restrict__ xi_f) {
  int idx = blockIdx.x * blockDim.x + threadIdx.x;
  if (idx >= NTOK * DINNER) return;
  const int tok = idx / DINNER;
  const int d   = idx - tok * DINNER;
  const int b   = tok / SEQ;
  const int s   = tok - b * SEQ;
  float acc = cb[d];
#pragma unroll
  for (int j = 0; j < 4; ++j) {
    int sp = s - 3 + j;
    if (sp >= 0)
      acc += cw[d * 4 + j] * xz[(size_t)(b * SEQ + sp) * (2 * DINNER) + d];
  }
  xi_f[idx] = acc / (1.f + __expf(-acc));        // SiLU
}

// dbc[tok, 0:80] = xi[tok,:] @ x_proj_w^T
__global__ void xproj_kernel(const float* __restrict__ xi_f,
                             const float* __restrict__ w,
                             float* __restrict__ dbc) {
  int idx = blockIdx.x * blockDim.x + threadIdx.x;
  if (idx >= NTOK * (DTRANK + 2 * DSTATE)) return;
  const int tok = idx / (DTRANK + 2 * DSTATE);
  const int j   = idx - tok * (DTRANK + 2 * DSTATE);
  const float* xr = xi_f + (size_t)tok * DINNER;
  const float* wr = w + (size_t)j * DINNER;
  float s = 0.f;
  for (int k = 0; k < DINNER; ++k) s += xr[k] * wr[k];
  dbc[idx] = s;
}

// FiLM gains/biases from context c: film[mat][batch][32]
__global__ void film_kernel(const float* __restrict__ c,
                            const float* __restrict__ Bw, const float* __restrict__ Bb,
                            const float* __restrict__ Cw, const float* __restrict__ Cb,
                            float* __restrict__ film) {
  const int t   = threadIdx.x;          // 0..127
  const int mat = t >> 6;
  const int b   = (t >> 5) & 1;
  const int j   = t & 31;
  const float* w    = (mat ? Cw : Bw) + (size_t)j * DMODEL;
  const float* bias = (mat ? Cb : Bb);
  const float* cr   = c + (size_t)b * DMODEL;
  float s = bias[j];
  for (int k = 0; k < DMODEL; ++k) s += cr[k] * w[k];
  film[(mat * 2 + b) * 32 + j] = s;
}

// delta = softplus(dbc[:, :48] @ dt_proj_w^T + dt_proj_b)
__global__ void delta_kernel(const float* __restrict__ dbc,
                             const float* __restrict__ w, const float* __restrict__ bias,
                             float* __restrict__ delta) {
  int idx = blockIdx.x * blockDim.x + threadIdx.x;
  if (idx >= NTOK * DINNER) return;
  const int tok = idx / DINNER;
  const int d   = idx - tok * DINNER;
  const float* dr = dbc + (size_t)tok * (DTRANK + 2 * DSTATE);
  const float* wr = w + (size_t)d * DTRANK;
  float s = bias[d];
#pragma unroll 8
  for (int r = 0; r < DTRANK; ++r) s += dr[r] * wr[r];
  delta[idx] = (s > 20.f) ? s : log1pf(__expf(s));
}

// Bm/Cm with FiLM applied
__global__ void bc_film_kernel(const float* __restrict__ dbc,
                               const float* __restrict__ film,
                               float* __restrict__ Bm, float* __restrict__ Cm) {
  int idx = blockIdx.x * blockDim.x + threadIdx.x;
  if (idx >= NTOK * 2 * DSTATE) return;
  const int tok = idx >> 5;
  const int j   = idx & 31;
  const int isC = j >> 4;
  const int n   = j & 15;
  const int b   = tok / SEQ;
  const float g  = film[(isC * 2 + b) * 32 + n];
  const float bb = film[(isC * 2 + b) * 32 + 16 + n];
  const float v  = dbc[(size_t)tok * (DTRANK + 2 * DSTATE) + DTRANK + isC * DSTATE + n];
  float* dst = isC ? Cm : Bm;
  dst[(size_t)tok * DSTATE + n] = v * g + bb;
}

// Selective scan: one lane per (channel, state); 16-lane shfl reduction for y.
// delta/x loaded once per channel (lane n==0) and shfl-broadcast.
// Epilogue fuses +D*x, SiLU(z) gating, bf16 store for the out_proj GEMM.
__global__ __launch_bounds__(256)
void scan_kernel(const float* __restrict__ delta, const float* __restrict__ xi_f,
                 const float* __restrict__ Bm,    const float* __restrict__ Cm,
                 const float* __restrict__ xz,    const float* __restrict__ Alog,
                 const float* __restrict__ Dv,    unsigned short* __restrict__ ys) {
  const int n = threadIdx.x & 15;                       // state index
  const int d = blockIdx.x * 16 + (threadIdx.x >> 4);   // channel
  const int b = blockIdx.y;                             // batch
  const float An = -__expf(Alog[(size_t)d * DSTATE + n]);
  const float Dd = Dv[d];
  float hst = 0.f;
  for (int t = 0; t < SEQ; ++t) {
    const int tok = b * SEQ + t;
    float dt = 0.f, x = 0.f;
    if (n == 0) {
      dt = delta[(size_t)tok * DINNER + d];
      x  = xi_f[(size_t)tok * DINNER + d];
    }
    dt = __shfl(dt, 0, 16);                             // broadcast in 16-group
    x  = __shfl(x, 0, 16);
    const float Bv = Bm[(size_t)tok * DSTATE + n];
    const float Cv = Cm[(size_t)tok * DSTATE + n];
    hst = __expf(dt * An) * hst + (dt * x) * Bv;
    float y = hst * Cv;
    y += __shfl_xor(y, 1, 16);
    y += __shfl_xor(y, 2, 16);
    y += __shfl_xor(y, 4, 16);
    y += __shfl_xor(y, 8, 16);
    if (n == 0) {
      const float yv = y + Dd * x;
      const float z  = xz[(size_t)tok * (2 * DINNER) + DINNER + d];
      const float sz = z / (1.f + __expf(-z));
      ys[(size_t)tok * DINNER + d] = f2bf(yv * sz);
    }
  }
}

// ---------------------------------------------------------------- launch
extern "C" void kernel_launch(void* const* d_in, const int* in_sizes, int n_in,
                              void* d_out, int out_size, void* d_ws, size_t ws_size,
                              hipStream_t stream) {
  const float* x        = (const float*)d_in[0];
  const float* c        = (const float*)d_in[1];
  const float* norm_w   = (const float*)d_in[2];
  const float* in_proj  = (const float*)d_in[3];
  const float* ctxB_w   = (const float*)d_in[4];
  const float* ctxB_b   = (const float*)d_in[5];
  const float* ctxC_w   = (const float*)d_in[6];
  const float* ctxC_b   = (const float*)d_in[7];
  const float* conv_w   = (const float*)d_in[8];
  const float* conv_b   = (const float*)d_in[9];
  const float* x_proj_w = (const float*)d_in[10];
  const float* dt_w     = (const float*)d_in[11];
  const float* dt_b     = (const float*)d_in[12];
  const float* A_log    = (const float*)d_in[13];
  const float* Dv       = (const float*)d_in[14];
  const float* out_proj = (const float*)d_in[15];
  float* h = (float*)d_out;

  // carve workspace
  size_t off = 0;
  auto carve = [&](size_t bytes) -> void* {
    void* p = (char*)d_ws + off;
    off += (bytes + 255) & ~(size_t)255;
    return p;
  };
  unsigned short* inw_bf  = (unsigned short*)carve((size_t)NLAYERS * 2 * DINNER * DMODEL * 2);
  unsigned short* outw_bf = (unsigned short*)carve((size_t)NLAYERS * DMODEL * DINNER * 2);
  unsigned short* xn_bf   = (unsigned short*)carve((size_t)NTOK * DMODEL * 2);
  float*          xzb     = (float*)carve((size_t)NTOK * 2 * DINNER * 4);
  float*          xi_f    = (float*)carve((size_t)NTOK * DINNER * 4);
  float*          dbc     = (float*)carve((size_t)NTOK * (DTRANK + 2 * DSTATE) * 4);
  float*          deltab  = (float*)carve((size_t)NTOK * DINNER * 4);
  float*          Bmb     = (float*)carve((size_t)NTOK * DSTATE * 4);
  float*          Cmb     = (float*)carve((size_t)NTOK * DSTATE * 4);
  float*          filmb   = (float*)carve(128 * 4);
  unsigned short* ys_bf   = (unsigned short*)carve((size_t)NTOK * DINNER * 2);
  (void)in_sizes; (void)n_in; (void)out_size; (void)ws_size;

  // h <- x (residual stream lives in d_out)
  hipMemcpyAsync(h, x, (size_t)NTOK * DMODEL * sizeof(float),
                 hipMemcpyDeviceToDevice, stream);

  // weight conversion f32 -> bf16 (all layers)
  {
    int n1 = NLAYERS * 2 * DINNER * DMODEL;
    cvt_f32_to_bf16<<<(n1 + 255) / 256, 256, 0, stream>>>(in_proj, inw_bf, n1);
    int n2 = NLAYERS * DMODEL * DINNER;
    cvt_f32_to_bf16<<<(n2 + 255) / 256, 256, 0, stream>>>(out_proj, outw_bf, n2);
  }

  for (int L = 0; L < NLAYERS; ++L) {
    const unsigned short* inwL  = inw_bf  + (size_t)L * 2 * DINNER * DMODEL;
    const unsigned short* outwL = outw_bf + (size_t)L * DMODEL * DINNER;

    // 1) RMSNorm -> bf16
    rmsnorm_kernel<<<NTOK, 256, 0, stream>>>(h, norm_w + (size_t)L * DMODEL, xn_bf);

    // 2) in_proj GEMM: [4096,768] x [3072,768]^T -> xz  (WMMA bf16, 32x64 tiles)
    {
      int waves = (NTOK / 32) * ((2 * DINNER) / 64);
      wmma_gemm_bf16<<<waves / 8, 256, 0, stream>>>(xn_bf, inwL, xzb, nullptr,
                                                    2 * DINNER, DMODEL);
    }

    // 3) depthwise conv + SiLU
    {
      int n = NTOK * DINNER;
      conv_silu_kernel<<<(n + 255) / 256, 256, 0, stream>>>(
          xzb, conv_w + (size_t)L * DINNER * 4, conv_b + (size_t)L * DINNER, xi_f);
    }

    // 4) x_proj -> dbc [4096,80]
    {
      int n = NTOK * (DTRANK + 2 * DSTATE);
      xproj_kernel<<<(n + 255) / 256, 256, 0, stream>>>(
          xi_f, x_proj_w + (size_t)L * (DTRANK + 2 * DSTATE) * DINNER, dbc);
    }

    // 5) FiLM params from context
    film_kernel<<<1, 128, 0, stream>>>(
        c, ctxB_w + (size_t)L * 2 * DSTATE * DMODEL, ctxB_b + (size_t)L * 2 * DSTATE,
        ctxC_w + (size_t)L * 2 * DSTATE * DMODEL, ctxC_b + (size_t)L * 2 * DSTATE, filmb);

    // 6) delta = softplus(dbc @ dt_proj^T + b)
    {
      int n = NTOK * DINNER;
      delta_kernel<<<(n + 255) / 256, 256, 0, stream>>>(
          dbc, dt_w + (size_t)L * DINNER * DTRANK, dt_b + (size_t)L * DINNER, deltab);
    }

    // 7) FiLM'd B/C matrices
    {
      int n = NTOK * 2 * DSTATE;
      bc_film_kernel<<<(n + 255) / 256, 256, 0, stream>>>(dbc, filmb, Bmb, Cmb);
    }

    // 8) selective scan (fused +D*x, SiLU(z) gate, bf16 store)
    {
      dim3 grid(DINNER / 16, BATCH);
      scan_kernel<<<grid, 256, 0, stream>>>(deltab, xi_f, Bmb, Cmb, xzb,
                                            A_log + (size_t)L * DINNER * DSTATE,
                                            Dv + (size_t)L * DINNER, ys_bf);
    }

    // 9) out_proj GEMM with residual: h += ys @ out_proj^T  (WMMA bf16)
    {
      int waves = (NTOK / 32) * (DMODEL / 64);
      wmma_gemm_bf16<<<waves / 8, 256, 0, stream>>>(ys_bf, outwL, h, h,
                                                    DMODEL, DINNER);
    }
  }
}
